// TransformerModel_70128226009220
// MI455X (gfx1250) — compile-verified, matrix operand order checked by reading, not run on hardware
//
#include <hip/hip_runtime.h>
#include <hip/hip_bf16.h>
#include <math.h>

// ---- Problem constants (must match reference) ----
#define Bb   2
#define Ss   16384
#define Dd   256
#define Hh   8
#define HD   32
#define Ll   4
#define DFF  1024
#define CS   512
#define NLm  32
#define NC   32          // S / CS
#define KVt  544         // NL + CS
#define EPSf 1e-5f

typedef __attribute__((ext_vector_type(16))) _Float16 v16h;
typedef __attribute__((ext_vector_type(8)))  _Float16 v8h;
typedef __attribute__((ext_vector_type(8)))  float    v8f;

// ---------------------------------------------------------------------------
// WMMA fragment helpers (wave32, V_WMMA_F32_16X16X32_F16)
// A-frag (16x32, row m = lane&15, lane group g = lane>>4 selects K halves):
//   element e: K = (e/8)*16 + (e%8) + 8*g  -> two contiguous 8-half (16B) loads
// B-frag identical pattern over K for fixed column n = lane&15.
// C-frag: lane n = lane&15, VGPR r -> row m = r + 8*(lane>>4).
// ---------------------------------------------------------------------------
__device__ __forceinline__ v16h frag_from_ptr(const _Float16* p) {
  v8h lo = *(const v8h*)p;
  v8h hi = *(const v8h*)(p + 16);
  v16h r;
#pragma unroll
  for (int i = 0; i < 8; ++i) { r[i] = lo[i]; r[i + 8] = hi[i]; }
  return r;
}

__device__ __forceinline__ v16h load_frag16x32(const _Float16* __restrict__ base,
                                               int stride, int r0, int k0, int lane) {
  const int m = lane & 15, g = lane >> 4;
  return frag_from_ptr(base + (size_t)(r0 + m) * (size_t)stride + k0 + 8 * g);
}

__device__ __forceinline__ v8f wmma_f16(v16h a, v16h b, v8f c) {
  return __builtin_amdgcn_wmma_f32_16x16x32_f16(false, a, false, b, (short)0, c,
                                                false, false);
}

// ---------------------------------------------------------------------------
// Generic GEMM: C[M,N] = A[M,K](f16) @ Wt[N,K](f16)^T + bias
// MODE 0: out f16   MODE 1: out f16 with exact GELU   MODE 2: f32 += (residual)
// blockDim = 256 (8 waves). Block computes 128x64; waves share a 64x32 weight
// sub-tile staged in LDS via GLOBAL_LOAD_ASYNC_TO_LDS_B128 (double-buffered,
// ASYNCcnt-synchronized). Requires M % 128 == 0, N % 64 == 0, K % 32 == 0.
// ---------------------------------------------------------------------------
template <int MODE>
__global__ void gemm_wmma_kernel(const _Float16* __restrict__ A,
                                 const _Float16* __restrict__ Wt,
                                 const float* __restrict__ bias,
                                 _Float16* __restrict__ outH,
                                 float* __restrict__ resF,
                                 int M, int N, int K) {
  __shared__ _Float16 tileB[2][64 * 32];  // two 4KB buffers, row stride 32 halves

  const int tid  = threadIdx.x;  // 0..255
  const int lane = tid & 31;
  const int wave = tid >> 5;     // 0..7
  const int col0 = blockIdx.x * 64;
  const int row0 = blockIdx.y * 128 + wave * 16;

  // async staging: thread t copies 16B: tile row = t/4 (0..63), segment = t%4
  const int srow = tid >> 2, sseg = tid & 3;
  const _Float16* gsrc = Wt + (size_t)(col0 + srow) * (size_t)K + sseg * 8;

  {  // stage k-slice 0 into buffer 0
    unsigned dst = (unsigned)(size_t)&tileB[0][srow * 32 + sseg * 8];
    asm volatile("global_load_async_to_lds_b128 %0, %1, off"
                 :: "v"(dst), "v"(gsrc) : "memory");
  }

  v8f acc0 = {}, acc1 = {}, acc2 = {}, acc3 = {};
  const int n = lane & 15, g = lane >> 4;
  const int nSteps = K >> 5;

  for (int it = 0; it < nSteps; ++it) {
    asm volatile("s_wait_asynccnt 0x0" ::: "memory");  // own stage complete
    __syncthreads();                                   // whole tile visible; prior reads done
    const int cur = it & 1;
    if (it + 1 < nSteps) {  // prefetch next k-slice into the other buffer
      unsigned dst = (unsigned)(size_t)&tileB[cur ^ 1][srow * 32 + sseg * 8];
      asm volatile("global_load_async_to_lds_b128 %0, %1, off"
                   :: "v"(dst), "v"(gsrc + (size_t)(it + 1) * 32) : "memory");
    }
    const int k0 = it * 32;
    v16h a = load_frag16x32(A, K, row0, k0, lane);
    const _Float16* tb = &tileB[cur][0];
    v16h b0 = frag_from_ptr(tb + (size_t)(n)      * 32 + 8 * g);
    v16h b1 = frag_from_ptr(tb + (size_t)(16 + n) * 32 + 8 * g);
    v16h b2 = frag_from_ptr(tb + (size_t)(32 + n) * 32 + 8 * g);
    v16h b3 = frag_from_ptr(tb + (size_t)(48 + n) * 32 + 8 * g);
    acc0 = wmma_f16(a, b0, acc0);
    acc1 = wmma_f16(a, b1, acc1);
    acc2 = wmma_f16(a, b2, acc2);
    acc3 = wmma_f16(a, b3, acc3);
  }

  const int hb = g;
#pragma unroll
  for (int j = 0; j < 4; ++j) {
    v8f acc = (j == 0) ? acc0 : (j == 1) ? acc1 : (j == 2) ? acc2 : acc3;
    const int gn = col0 + 16 * j + n;
    const float bv = bias[gn];
#pragma unroll
    for (int r = 0; r < 8; ++r) {
      const int gm = row0 + r + 8 * hb;
      float v = acc[r] + bv;
      if (MODE == 1) v = 0.5f * v * (1.0f + erff(v * 0.70710678118654752f));
      if (MODE == 2) resF[(size_t)gm * N + gn] += v;
      else           outH[(size_t)gm * N + gn] = (_Float16)v;
    }
  }
}

// ---------------------------------------------------------------------------
// Chunked attention: one wave handles 16 query rows of one (batch, chunk, head).
// scores (544 cols) via WMMA K=32 (head dim), softmax in LDS, PV via WMMA with
// f16 probs re-loaded as A-fragments straight from LDS.
// grid = (CS/16, NC, B*H), blockDim = 32.
// ---------------------------------------------------------------------------
__global__ void attn_kernel(const _Float16* __restrict__ qb,
                            const _Float16* __restrict__ kx,
                            const _Float16* __restrict__ klm,
                            const _Float16* __restrict__ vx,
                            const _Float16* __restrict__ vlm,
                            _Float16* __restrict__ ao) {
  __shared__ float    sc[16 * KVt];   // raw scores
  __shared__ _Float16 pb[16 * KVt];   // normalized probs (f16)

  const int lane = threadIdx.x;
  const int row0 = blockIdx.x * 16;
  const int ch   = blockIdx.y;
  const int bh   = blockIdx.z;
  const int b    = bh >> 3;
  const int hh   = bh & 7;
  const size_t chunkRow = (size_t)b * Ss + (size_t)ch * CS;  // row in [B*S]
  const float scale = 0.17677669529663687f;                  // 1/sqrt(32)

  // Q fragment: 16 rows x 32 (full head dim)
  const _Float16* qbase = qb + chunkRow * Dd + hh * HD;
  v16h aq = load_frag16x32(qbase, Dd, row0, 0, lane);

  const int n = lane & 15, g = lane >> 4;

  // ---- scores: 34 tiles of 16 columns ----
  for (int c = 0; c < KVt; c += 16) {
    const int tok = c + n;
    const _Float16* kp = (tok < NLm)
        ? (klm + ((size_t)b * NLm + tok) * Dd + hh * HD)
        : (kx + (chunkRow + (tok - NLm)) * Dd + hh * HD);
    v16h bk = frag_from_ptr(kp + 8 * g);
    v8f s = {};
    s = wmma_f16(aq, bk, s);
#pragma unroll
    for (int r = 0; r < 8; ++r) sc[(r + 8 * g) * KVt + c + n] = s[r] * scale;
  }
  __syncthreads();

  // ---- softmax: lanes 0..15 each own one row ----
  if (lane < 16) {
    const float* rowp = sc + lane * KVt;
    _Float16*    prow = pb + lane * KVt;
    float mx = -3.402823466e38f;
    for (int j = 0; j < KVt; ++j) mx = fmaxf(mx, rowp[j]);
    float sum = 0.f;
    for (int j = 0; j < KVt; ++j) {
      float e = expf(rowp[j] - mx);
      sum += e;
      prow[j] = (_Float16)e;
    }
    const float inv = 1.0f / sum;
    for (int j = 0; j < KVt; ++j) prow[j] = (_Float16)((float)prow[j] * inv);
  }
  __syncthreads();

  // ---- PV: 17 K-steps of 32 tokens, 2 output tiles (head dim 32) ----
  v8f o0 = {}, o1 = {};
  const _Float16* prow = pb + (size_t)(lane & 15) * KVt;
  for (int c = 0; c < KVt; c += 32) {
    v16h ap = frag_from_ptr(prow + c + 8 * g);
    v16h b0, b1;
#pragma unroll
    for (int e = 0; e < 16; ++e) {
      const int kk  = ((e >> 3) << 4) + (e & 7) + 8 * g;
      const int tok = c + kk;
      const _Float16* vp = (tok < NLm)
          ? (vlm + ((size_t)b * NLm + tok) * Dd + hh * HD)
          : (vx + (chunkRow + (tok - NLm)) * Dd + hh * HD);
      b0[e] = vp[n];
      b1[e] = vp[16 + n];
    }
    o0 = wmma_f16(ap, b0, o0);
    o1 = wmma_f16(ap, b1, o1);
  }

  _Float16* obase = ao + (chunkRow + row0) * Dd + hh * HD;
#pragma unroll
  for (int r = 0; r < 8; ++r) {
    const int m = r + 8 * g;
    obase[(size_t)m * Dd + n]      = (_Float16)o0[r];
    obase[(size_t)m * Dd + 16 + n] = (_Float16)o1[r];
  }
}

// ---------------------------------------------------------------------------
// Small glue kernels
// ---------------------------------------------------------------------------
__global__ void embed_pe_kernel(const int* __restrict__ x,
                                const float* __restrict__ emb,
                                float* __restrict__ h) {
  const size_t i = (size_t)blockIdx.x * 256 + threadIdx.x;  // B*S*D
  const int d = (int)(i & 255);
  const size_t bs = i >> 8;
  const int s = (int)(bs % Ss);
  const int tok = x[bs];
  const float div = expf(-logf(10000.0f) * (float)(2 * (d >> 1)) / 256.0f);
  const float ang = (float)s * div;
  const float pe = (d & 1) ? cosf(ang) : sinf(ang);
  h[i] = emb[(size_t)tok * Dd + d] + pe;
}

// wave per row; blockDim=256 (8 waves -> 8 rows/block)
__global__ void layernorm_kernel(const float* __restrict__ hin,
                                 const float* __restrict__ gg,
                                 const float* __restrict__ be,
                                 _Float16* __restrict__ out) {
  const int lane = threadIdx.x & 31, w = threadIdx.x >> 5;
  const size_t row = (size_t)blockIdx.x * 8 + w;
  const float* p = hin + row * Dd;
  float vals[8], s = 0.f, s2 = 0.f;
#pragma unroll
  for (int i = 0; i < 8; ++i) {
    float v = p[lane + 32 * i];
    vals[i] = v; s += v; s2 += v * v;
  }
  for (int off = 16; off; off >>= 1) {
    s  += __shfl_down(s,  off, 32);
    s2 += __shfl_down(s2, off, 32);
  }
  s  = __shfl(s, 0, 32);
  s2 = __shfl(s2, 0, 32);
  const float mean = s * (1.0f / Dd);
  const float var  = s2 * (1.0f / Dd) - mean * mean;
  const float inv  = rsqrtf(var + EPSf);
  _Float16* o = out + row * Dd;
#pragma unroll
  for (int i = 0; i < 8; ++i) {
    const int d = lane + 32 * i;
    o[d] = (_Float16)((vals[i] - mean) * inv * gg[d] + be[d]);
  }
}

__global__ void landmark_kernel(const _Float16* __restrict__ ln,
                                _Float16* __restrict__ lm) {
  const int i = blockIdx.x * 256 + threadIdx.x;  // B*NL*D = 16384
  const int d  = i & 255;
  const int nl = (i >> 8) & 31;
  const int b  = i >> 13;
  const _Float16* p = ln + ((size_t)b * Ss + (size_t)nl * 512) * Dd + d;
  float s = 0.f;
  for (int j = 0; j < 512; ++j) s += (float)p[(size_t)j * Dd];
  lm[(size_t)(b * NLm + nl) * Dd + d] = (_Float16)(s * (1.0f / 512.0f));
}

// mean over sequence: wave per (b,d); blockDim=256 -> 8 outputs/block
__global__ void pool_kernel(const float* __restrict__ h, float* __restrict__ pooled) {
  const int lane = threadIdx.x & 31, w = threadIdx.x >> 5;
  const int idx = blockIdx.x * 8 + w;      // b*D + d, total 512
  const int d = idx & 255, b = idx >> 8;
  const float* p = h + (size_t)b * Ss * Dd + d;
  float s = 0.f;
  for (int j = lane; j < Ss; j += 32) s += p[(size_t)j * Dd];
  for (int off = 16; off; off >>= 1) s += __shfl_down(s, off, 32);
  if (lane == 0) pooled[idx] = s * (1.0f / Ss);
}

// classifier head: one block per batch element, 128 threads
__global__ void head_kernel(const float* __restrict__ pooled,
                            const float* __restrict__ gh, const float* __restrict__ bh,
                            const float* __restrict__ Wh1, const float* __restrict__ bh1,
                            const float* __restrict__ Wh2, const float* __restrict__ bh2,
                            float* __restrict__ out) {
  __shared__ float xn[Dd];
  __shared__ float hid[128];
  __shared__ float red[2];
  const int b = blockIdx.x, t = threadIdx.x;
  if (t == 0) {
    float s = 0.f, s2 = 0.f;
    for (int d = 0; d < Dd; ++d) {
      float v = pooled[b * Dd + d];
      s += v; s2 += v * v;
    }
    const float mean = s * (1.0f / Dd);
    const float var  = s2 * (1.0f / Dd) - mean * mean;
    red[0] = mean;
    red[1] = rsqrtf(var + EPSf);
  }
  __syncthreads();
  for (int d = t; d < Dd; d += 128)
    xn[d] = (pooled[b * Dd + d] - red[0]) * red[1] * gh[d] + bh[d];
  __syncthreads();
  float a = bh1[t];
  for (int d = 0; d < Dd; ++d) a += xn[d] * Wh1[d * 128 + t];
  a = 0.5f * a * (1.0f + erff(a * 0.70710678118654752f));
  hid[t] = a * Wh2[t];
  __syncthreads();
  if (t == 0) {
    float s = bh2[0];
    for (int j = 0; j < 128; ++j) s += hid[j];
    out[b] = s;
  }
}

// convert f32 weight [K,N] -> f16 transposed [N,K]
__global__ void wtrans_kernel(const float* __restrict__ W, _Float16* __restrict__ Wt,
                              int K, int N) {
  const size_t i = (size_t)blockIdx.x * 256 + threadIdx.x;
  if (i >= (size_t)K * N) return;
  const int k = (int)(i % K);
  const int nn = (int)(i / K);
  Wt[i] = (_Float16)W[(size_t)k * N + nn];
}

// ---------------------------------------------------------------------------
// Host launcher
// ---------------------------------------------------------------------------
extern "C" void kernel_launch(void* const* d_in, const int* in_sizes, int n_in,
                              void* d_out, int out_size, void* d_ws, size_t ws_size,
                              hipStream_t stream) {
  (void)in_sizes; (void)n_in; (void)out_size; (void)ws_size;
  const int*   x_   = (const int*)  d_in[0];
  const float* emb  = (const float*)d_in[1];
  const float* Wq   = (const float*)d_in[2];
  const float* bq   = (const float*)d_in[3];
  const float* Wk   = (const float*)d_in[4];
  const float* bk   = (const float*)d_in[5];
  const float* Wv   = (const float*)d_in[6];
  const float* bv   = (const float*)d_in[7];
  const float* Wo   = (const float*)d_in[8];
  const float* bo   = (const float*)d_in[9];
  const float* g1   = (const float*)d_in[10];
  const float* be1  = (const float*)d_in[11];
  const float* g2   = (const float*)d_in[12];
  const float* be2  = (const float*)d_in[13];
  const float* W1   = (const float*)d_in[14];
  const float* b1   = (const float*)d_in[15];
  const float* W2   = (const float*)d_in[16];
  const float* b2   = (const float*)d_in[17];
  const float* gh   = (const float*)d_in[18];
  const float* bhp  = (const float*)d_in[19];
  const float* Wh1  = (const float*)d_in[20];
  const float* bh1  = (const float*)d_in[21];
  const float* Wh2  = (const float*)d_in[22];
  const float* bh2  = (const float*)d_in[23];
  float* out = (float*)d_out;

  // ---- workspace arena ----
  char* base = (char*)d_ws;
  size_t off = 0;
  auto alloc = [&](size_t bytes) { size_t o = off; off = (off + bytes + 255) & ~(size_t)255; return o; };
  const size_t nTok = (size_t)Bb * Ss;           // 32768 rows
  float*    h   = (float*)   (base + alloc(nTok * Dd * 4));      // 33.5 MB
  _Float16* ln  = (_Float16*)(base + alloc(nTok * Dd * 2));      // 16.8 MB
  // attention arena (4 x 16.8 MB); also aliased as MLP intermediate [B*S, DFF]
  _Float16* qb  = (_Float16*)(base + alloc(4 * nTok * Dd * 2));  // 67.1 MB
  _Float16* kx  = qb + nTok * Dd;
  _Float16* vx  = kx + nTok * Dd;
  _Float16* ao  = vx + nTok * Dd;
  _Float16* mb  = qb;                                            // [B*S, DFF] f16
  // landmark buffers padded to 128 rows (GEMM block tile M=128)
  _Float16* klm = (_Float16*)(base + alloc((size_t)128 * Dd * 2));
  _Float16* vlm = (_Float16*)(base + alloc((size_t)128 * Dd * 2));
  _Float16* lm  = (_Float16*)(base + alloc((size_t)128 * Dd * 2));
  float*    pooled = (float*)(base + alloc((size_t)Bb * Dd * 4));
  _Float16* wt  = (_Float16*)(base + alloc((size_t)Ll * 786432 * 2)); // 6.3 MB

  const dim3 blk256(256);
  const size_t lstride = 786432;  // f16 elems per layer in wt arena

  // ---- one-time weight convert + transpose (f32 [K,N] -> f16 [N,K]) ----
  for (int l = 0; l < Ll; ++l) {
    _Float16* wl = wt + (size_t)l * lstride;
    wtrans_kernel<<<256, 256, 0, stream>>>(Wq + (size_t)l * 65536, wl + 0,      Dd, Dd);
    wtrans_kernel<<<256, 256, 0, stream>>>(Wk + (size_t)l * 65536, wl + 65536,  Dd, Dd);
    wtrans_kernel<<<256, 256, 0, stream>>>(Wv + (size_t)l * 65536, wl + 131072, Dd, Dd);
    wtrans_kernel<<<256, 256, 0, stream>>>(Wo + (size_t)l * 65536, wl + 196608, Dd, Dd);
    wtrans_kernel<<<1024, 256, 0, stream>>>(W1 + (size_t)l * 262144, wl + 262144, Dd,  DFF);
    wtrans_kernel<<<1024, 256, 0, stream>>>(W2 + (size_t)l * 262144, wl + 524288, DFF, Dd);
  }

  // ---- embedding + positional encoding ----
  embed_pe_kernel<<<(unsigned)(nTok * Dd / 256), 256, 0, stream>>>(x_, emb, h);

  const int Mrows = (int)nTok;  // 32768
  for (int l = 0; l < Ll; ++l) {
    _Float16* wq_t = wt + (size_t)l * lstride + 0;
    _Float16* wk_t = wt + (size_t)l * lstride + 65536;
    _Float16* wv_t = wt + (size_t)l * lstride + 131072;
    _Float16* wo_t = wt + (size_t)l * lstride + 196608;
    _Float16* w1_t = wt + (size_t)l * lstride + 262144;
    _Float16* w2_t = wt + (size_t)l * lstride + 524288;
    const float* bq_l = bq + (size_t)l * Dd;
    const float* bk_l = bk + (size_t)l * Dd;
    const float* bv_l = bv + (size_t)l * Dd;
    const float* bo_l = bo + (size_t)l * Dd;
    const float* b1_l = b1 + (size_t)l * DFF;
    const float* b2_l = b2 + (size_t)l * Dd;

    // LN1 -> f16
    layernorm_kernel<<<Mrows / 8, 256, 0, stream>>>(h, g1 + (size_t)l * Dd, be1 + (size_t)l * Dd, ln);
    // landmarks (mean-pool 512-token segments of LN output)
    landmark_kernel<<<64, 256, 0, stream>>>(ln, lm);
    // projections (8-wave blocks, 128x64 tiles, async-LDS-staged weights)
    gemm_wmma_kernel<0><<<dim3(Dd / 64, Mrows / 128), blk256, 0, stream>>>(ln, wq_t, bq_l, qb, nullptr, Mrows, Dd, Dd);
    gemm_wmma_kernel<0><<<dim3(Dd / 64, Mrows / 128), blk256, 0, stream>>>(ln, wk_t, bk_l, kx, nullptr, Mrows, Dd, Dd);
    gemm_wmma_kernel<0><<<dim3(Dd / 64, 1),           blk256, 0, stream>>>(lm, wk_t, bk_l, klm, nullptr, 128, Dd, Dd);
    gemm_wmma_kernel<0><<<dim3(Dd / 64, Mrows / 128), blk256, 0, stream>>>(ln, wv_t, bv_l, vx, nullptr, Mrows, Dd, Dd);
    gemm_wmma_kernel<0><<<dim3(Dd / 64, 1),           blk256, 0, stream>>>(lm, wv_t, bv_l, vlm, nullptr, 128, Dd, Dd);
    // attention
    attn_kernel<<<dim3(CS / 16, NC, Bb * Hh), dim3(32), 0, stream>>>(qb, kx, klm, vx, vlm, ao);
    // output projection with residual add into h (f32)
    gemm_wmma_kernel<2><<<dim3(Dd / 64, Mrows / 128), blk256, 0, stream>>>(ao, wo_t, bo_l, nullptr, h, Mrows, Dd, Dd);
    // LN2 -> f16
    layernorm_kernel<<<Mrows / 8, 256, 0, stream>>>(h, g2 + (size_t)l * Dd, be2 + (size_t)l * Dd, ln);
    // MLP: GELU GEMM then residual GEMM
    gemm_wmma_kernel<1><<<dim3(DFF / 64, Mrows / 128), blk256, 0, stream>>>(ln, w1_t, b1_l, mb, nullptr, Mrows, DFF, Dd);
    gemm_wmma_kernel<2><<<dim3(Dd / 64, Mrows / 128), blk256, 0, stream>>>(mb, w2_t, b2_l, nullptr, h, Mrows, Dd, DFF);
  }

  // ---- pool + head ----
  pool_kernel<<<(Bb * Dd) / 8, 256, 0, stream>>>(h, pooled);
  head_kernel<<<Bb, 128, 0, stream>>>(pooled, gh, bhp, Wh1, bh1, Wh2, bh2, out);
}
